// MULTIGCN_17901423690508
// MI455X (gfx1250) — compile-verified, matrix operand order checked by reading, not run on hardware
//
#include <hip/hip_runtime.h>
#include <hip/hip_bf16.h>

// ---------------------------------------------------------------------------
// Multi-branch GCN for MI455X (gfx1250): bf16 WMMA GEMMs + atomic SpMM.
// ---------------------------------------------------------------------------

typedef __attribute__((ext_vector_type(16))) __bf16 v16bf;
typedef __attribute__((ext_vector_type(8)))  float  v8f;

constexpr int DIM      = 25;
constexpr int N_NODES  = 20000;     // multiple of 16
constexpr int N_EDGES  = 320000;
constexpr int NFEAT    = 300;
constexpr int NFEAT_P  = 320;       // pad K to multiple of 32
constexpr int NHID     = 64;
constexpr int NCLASS   = 50;
constexpr int NCLASS_P = 64;        // pad N to multiple of 16 (zero rows in W2^T)
constexpr int K2       = DIM * NHID; // 1600, already multiple of 32

// ---------------------------- utility kernels ------------------------------

__global__ void zero_f32_kernel(float* __restrict__ p, long long n) {
  long long i = (long long)blockIdx.x * blockDim.x + threadIdx.x;
  if (i < n) p[i] = 0.0f;
}

// src [R x C] f32 (row-major) -> dst [R x Cpad] bf16, zero-padded columns.
__global__ void cvt_pad_bf16_kernel(const float* __restrict__ src, __bf16* __restrict__ dst,
                                    int R, int C, int Cpad) {
  long long i = (long long)blockIdx.x * blockDim.x + threadIdx.x;
  long long total = (long long)R * Cpad;
  if (i >= total) return;
  int r = (int)(i / Cpad);
  int c = (int)(i % Cpad);
  float v = (c < C) ? src[(long long)r * C + c] : 0.0f;
  dst[i] = (__bf16)v;
}

// src [D][K][N] f32 -> dst [D][Npad][Kpad] bf16 (transposed per branch, zero pad).
__global__ void wt_pad_bf16_kernel(const float* __restrict__ src, __bf16* __restrict__ dst,
                                   int D, int K, int Kpad, int N, int Npad) {
  long long i = (long long)blockIdx.x * blockDim.x + threadIdx.x;
  long long per = (long long)Npad * Kpad;
  long long total = (long long)D * per;
  if (i >= total) return;
  int d = (int)(i / per);
  long long rem = i % per;
  int n = (int)(rem / Kpad);
  int k = (int)(rem % Kpad);
  float v = 0.0f;
  if (k < K && n < N) v = src[((long long)d * K + k) * N + n];
  dst[i] = (__bf16)v;
}

// ---------------------------- WMMA GEMM ------------------------------------
// Fragment loader for V_WMMA_F32_16X16X32_BF16.
// A: row-major [M x ld]; lane L (0-15) holds row majorIdx+L, lanes 16-31 the same
// rows with the second K half-group. Per ISA 16-bit A 16x32 layout:
//   lanes 0-15 : elems 0..7 -> K = kb+0..7,  elems 8..15 -> K = kb+16..23
//   lanes 16-31: elems 0..7 -> K = kb+8..15, elems 8..15 -> K = kb+24..31
// B is supplied transposed (row-major [N x ld]) so columns load identically.
__device__ __forceinline__ v16bf load_frag_bf16(const __bf16* __restrict__ base, int ld,
                                                int majorIdx, int kb, int lane) {
  const __bf16* p = base + (long long)(majorIdx + (lane & 15)) * ld + kb + ((lane & 16) ? 8 : 0);
  union { uint4 u[2]; v16bf v; } f;
  f.u[0] = *reinterpret_cast<const uint4*>(p);        // elems 0..7  (16 B)
  f.u[1] = *reinterpret_cast<const uint4*>(p + 16);   // elems 8..15 (16 B)
  return f.v;
}

// C[M x ldc] f32 = A[M x Kpad] bf16 * Bt[N x Kpad]^T bf16.
// grid.x = M/16, grid.y = N/64, block = 128 (4 waves, one 16-wide N tile each).
__global__ void __launch_bounds__(128)
wmma_gemm_bf16_kernel(const __bf16* __restrict__ A, const __bf16* __restrict__ Bt,
                      float* __restrict__ C, int Kpad, int ldc) {
  const int lane = threadIdx.x & 31;
  const int wave = threadIdx.x >> 5;
  const int row  = blockIdx.x * 16;
  const int col  = (blockIdx.y * 4 + wave) * 16;

  v8f acc = {};
  for (int kb = 0; kb < Kpad; kb += 32) {
    v16bf a = load_frag_bf16(A,  Kpad, row, kb, lane);
    v16bf b = load_frag_bf16(Bt, Kpad, col, kb, lane);
    acc = __builtin_amdgcn_wmma_f32_16x16x32_bf16(
        /*neg_a=*/false, a, /*neg_b=*/false, b,
        /*c_mod=*/(short)0, acc, /*reuse_a=*/false, /*reuse_b=*/false);
  }

  // C layout: acc[i] -> (row + i + (lane<16?0:8), col + (lane&15))
  const int cr = row + ((lane & 16) ? 8 : 0);
  const int cc = col + (lane & 15);
#pragma unroll
  for (int i = 0; i < 8; ++i)
    C[(long long)(cr + i) * ldc + cc] = acc[i];
}

// ---------------------------- SpMM scatter ---------------------------------
// One wave per edge; lane handles features lane and lane+32 (F fixed at 64).
// dst[row*dstLd + dstOff + f] += val * src[col*64 + f]
__global__ void __launch_bounds__(256)
spmm_scatter64_kernel(const int* __restrict__ rows, const int* __restrict__ cols,
                      const float* __restrict__ vals, const float* __restrict__ src,
                      float* __restrict__ dst, int nEdges, int dstLd, int dstOff) {
  long long t = (long long)blockIdx.x * blockDim.x + threadIdx.x;
  int e = (int)(t >> 5);
  if (e >= nEdges) return;
  int lane = (int)(t & 31);
  int r = rows[e];
  int c = cols[e];
  float v = vals[e];
  const float* s = src + (long long)c * 64;
  float* d = dst + (long long)r * dstLd + dstOff;
  atomicAdd(d + lane,      v * s[lane]);
  atomicAdd(d + lane + 32, v * s[lane + 32]);
}

// h1cat (fp32, [N_NODES x K2]) + b1 (flat [DIM*NHID] == column index) -> relu -> bf16
__global__ void bias_relu_cvt_kernel(const float* __restrict__ h, const float* __restrict__ b1,
                                     __bf16* __restrict__ out) {
  long long i = (long long)blockIdx.x * blockDim.x + threadIdx.x;
  long long total = (long long)N_NODES * K2;
  if (i >= total) return;
  int c = (int)(i % K2);
  float v = h[i] + b1[c];
  out[i] = (__bf16)fmaxf(v, 0.0f);
}

// out[n,c] = max(out[n,c], relu(acc[n*64 + c] + b2[c])) for c < NCLASS
__global__ void relu_bias_max_kernel(const float* __restrict__ acc, const float* __restrict__ b2,
                                     float* __restrict__ out) {
  int i = blockIdx.x * blockDim.x + threadIdx.x;
  if (i >= N_NODES * NCLASS) return;
  int node = i / NCLASS;
  int c    = i % NCLASS;
  float v = fmaxf(acc[(long long)node * NCLASS_P + c] + b2[c], 0.0f);
  out[i] = fmaxf(out[i], v);
}

// ---------------------------- host launcher --------------------------------

static inline size_t align256(size_t x) { return (x + 255) & ~size_t(255); }
static inline int blocks_for(long long n, int b) { return (int)((n + b - 1) / b); }

extern "C" void kernel_launch(void* const* d_in, const int* in_sizes, int n_in,
                              void* d_out, int out_size, void* d_ws, size_t ws_size,
                              hipStream_t stream) {
  (void)in_sizes; (void)n_in; (void)out_size; (void)ws_size;

  const float* x     = (const float*)d_in[0];
  const int*   arows = (const int*)  d_in[1];
  const int*   acols = (const int*)  d_in[2];
  const float* avals = (const float*)d_in[3];
  const float* W1    = (const float*)d_in[4];
  const float* b1    = (const float*)d_in[5];
  const float* W2    = (const float*)d_in[6];
  const float* b2    = (const float*)d_in[7];
  float* out = (float*)d_out;

  // workspace suballocation (256B aligned; total ~222 MB)
  char* ws = (char*)d_ws;
  size_t off = 0;
  auto suballoc = [&](size_t bytes) { char* p = ws + off; off = align256(off + bytes); return p; };
  __bf16* xb  = (__bf16*)suballoc((size_t)N_NODES * NFEAT_P * sizeof(__bf16));   // 12.8 MB
  __bf16* w1t = (__bf16*)suballoc((size_t)DIM * NHID * NFEAT_P * sizeof(__bf16)); // 1.0 MB
  __bf16* w2t = (__bf16*)suballoc((size_t)DIM * NCLASS_P * K2 * sizeof(__bf16));  // 5.1 MB
  float*  h1  = (float*) suballoc((size_t)N_NODES * K2 * sizeof(float));          // 128 MB
  __bf16* h1b = (__bf16*)suballoc((size_t)N_NODES * K2 * sizeof(__bf16));         // 64 MB
  float*  sup = (float*) suballoc((size_t)N_NODES * 64 * sizeof(float));          // 5.1 MB (reused per branch)
  float*  h2  = (float*) suballoc((size_t)N_NODES * 64 * sizeof(float));          // 5.1 MB (reused per branch)

  // --- init accumulators ---
  {
    long long n1 = (long long)N_NODES * K2;
    zero_f32_kernel<<<blocks_for(n1, 256), 256, 0, stream>>>(h1, n1);
    long long no = (long long)N_NODES * NCLASS;
    zero_f32_kernel<<<blocks_for(no, 256), 256, 0, stream>>>(out, no);
  }

  // --- convert inputs to bf16 (padded / transposed) ---
  cvt_pad_bf16_kernel<<<blocks_for((long long)N_NODES * NFEAT_P, 256), 256, 0, stream>>>(
      x, xb, N_NODES, NFEAT, NFEAT_P);
  wt_pad_bf16_kernel<<<blocks_for((long long)DIM * NHID * NFEAT_P, 256), 256, 0, stream>>>(
      W1, w1t, DIM, NFEAT, NFEAT_P, NHID, NHID);
  wt_pad_bf16_kernel<<<blocks_for((long long)DIM * NCLASS_P * K2, 256), 256, 0, stream>>>(
      W2, w2t, DIM, K2, K2, NCLASS, NCLASS_P);

  const dim3 gemmGrid(N_NODES / 16, NHID / 64);      // (1250, 1)
  const int spmmBlocks = blocks_for((long long)N_EDGES * 32, 256);

  // --- layer 1: per branch, GEMM (WMMA) then SpMM scatter into h1cat columns ---
  for (int d = 0; d < DIM; ++d) {
    const __bf16* Bt = w1t + (size_t)d * NHID * NFEAT_P;
    wmma_gemm_bf16_kernel<<<gemmGrid, 128, 0, stream>>>(xb, Bt, sup, NFEAT_P, 64);
    spmm_scatter64_kernel<<<spmmBlocks, 256, 0, stream>>>(
        arows + (size_t)d * N_EDGES, acols + (size_t)d * N_EDGES,
        avals + (size_t)d * N_EDGES, sup, h1, N_EDGES, K2, d * NHID);
  }

  // --- bias + relu + convert to bf16 (layer-2 A matrix) ---
  bias_relu_cvt_kernel<<<blocks_for((long long)N_NODES * K2, 256), 256, 0, stream>>>(h1, b1, h1b);

  // --- layer 2: per branch, zero, GEMM (WMMA), SpMM scatter, relu+bias+max ---
  const long long nH2 = (long long)N_NODES * NCLASS_P;
  const int maxBlocks = blocks_for((long long)N_NODES * NCLASS, 256);
  for (int d = 0; d < DIM; ++d) {
    zero_f32_kernel<<<blocks_for(nH2, 256), 256, 0, stream>>>(h2, nH2);
    const __bf16* Bt = w2t + (size_t)d * NCLASS_P * K2;
    wmma_gemm_bf16_kernel<<<gemmGrid, 128, 0, stream>>>(h1b, Bt, sup, K2, 64);
    spmm_scatter64_kernel<<<spmmBlocks, 256, 0, stream>>>(
        arows + (size_t)d * N_EDGES, acols + (size_t)d * N_EDGES,
        avals + (size_t)d * N_EDGES, sup, h2, N_EDGES, NCLASS_P, 0);
    relu_bias_max_kernel<<<maxBlocks, 256, 0, stream>>>(h2, b2 + (size_t)d * NCLASS, out);
  }
}
